// Encoder_13795434955501
// MI455X (gfx1250) — matrix operand order repaired and checked by reading, not compiled, not run on hardware
//
#include <hip/hip_runtime.h>

// ---------------------------------------------------------------------------
// Types for WMMA
// ---------------------------------------------------------------------------
typedef __attribute__((ext_vector_type(16))) __bf16        v16bf;
typedef __attribute__((ext_vector_type(8)))  float         v8f;
typedef __attribute__((ext_vector_type(4)))  unsigned int  u32x4;

union FragB16 { u32x4 q[2]; v16bf v; };

#define D_MODEL 512
#define HEADS   8
#define DHEAD   64
#define DFF     2048
#define LAYERS  4
#define SEQ     4096
#define BATCH   4
#define MROWS   (BATCH * SEQ)   // 16384

__device__ __forceinline__ unsigned short f32_to_bf16(float f) {
    unsigned int u = __float_as_uint(f);
    u += 0x7fffu + ((u >> 16) & 1u);   // round-to-nearest-even
    return (unsigned short)(u >> 16);
}
__device__ __forceinline__ float bf16_to_f32(unsigned int h) {
    return __uint_as_float((h & 0xffffu) << 16);
}

// ---------------------------------------------------------------------------
// CDNA5 async Global -> LDS copy (ASYNCcnt path)
//   global_load_async_to_lds_b128  vdst(LDS byte addr), vaddr(64b), off
// LDS byte offset = low 32 bits of the flat pointer to a __shared__ object.
// ---------------------------------------------------------------------------
__device__ __forceinline__ void async_b128(unsigned lds_addr, const void* g) {
    asm volatile("global_load_async_to_lds_b128 %0, %1, off"
                 :: "v"(lds_addr), "v"((unsigned long long)(size_t)g)
                 : "memory");
}
__device__ __forceinline__ void wait_async_le6() {
    asm volatile("s_wait_asynccnt 0x6" ::: "memory");
}
__device__ __forceinline__ void wait_async_0() {
    asm volatile("s_wait_asynccnt 0x0" ::: "memory");
}

// ---------------------------------------------------------------------------
// fp32 [L,K,N] -> bf16 transposed [L,N,K]  (weights; done once per call)
// ---------------------------------------------------------------------------
__global__ __launch_bounds__(256) void cvtT_bf16_kernel(const float* __restrict__ in,
                                                        unsigned short* __restrict__ out,
                                                        int K, int N, size_t total) {
    size_t i = (size_t)blockIdx.x * 256 + threadIdx.x;
    if (i >= total) return;
    size_t per = (size_t)K * N;
    size_t l = i / per;
    int rem = (int)(i - l * per);
    int k = rem / N;
    int n = rem - k * N;
    out[l * per + (size_t)n * K + k] = f32_to_bf16(in[i]);
}

// ---------------------------------------------------------------------------
// Embedding gather * sqrt(d) + positional encoding -> x (f32) and x (bf16)
// ---------------------------------------------------------------------------
__global__ __launch_bounds__(256) void embed_kernel(const int* __restrict__ tokens,
                                                    const float* __restrict__ emb,
                                                    const float* __restrict__ pe,
                                                    float* __restrict__ xf,
                                                    unsigned short* __restrict__ xb) {
    size_t i = (size_t)blockIdx.x * 256 + threadIdx.x;   // over MROWS*D_MODEL
    int c = (int)(i & (D_MODEL - 1));
    size_t row = i >> 9;                                  // token row 0..16383
    int s = (int)(row & (SEQ - 1));                       // position in sequence
    int tok = tokens[row];
    float v = emb[(size_t)tok * D_MODEL + c] * 22.6274169979695f   // sqrt(512)
            + pe[(size_t)s * D_MODEL + c];
    xf[i] = v;
    xb[i] = f32_to_bf16(v);
}

// ---------------------------------------------------------------------------
// bf16 WMMA GEMM: C[M,N] = A[M,K] * Bt[N,K]^T + bias[N]  (A,Bt bf16; acc f32)
// Block: 256 threads = 8 waves; block tile 128x256; BK = 32.
// Each wave (wm 0..1, wn 0..3) owns a 64x64 region = 4x4 WMMA 16x16 tiles:
// 16 v_wmma per K-step against 16 ds_load_b128 (A,B fragments reused 4x).
// Double-buffered LDS, staged with async global->LDS b128 copies
// (6 per wave per stage; pipeline waits s_wait_asynccnt<=6).
// ---------------------------------------------------------------------------
template<int OUT_BF16, int RELU>
__global__ __launch_bounds__(256) void gemm_bf16_wmma(
    const unsigned short* __restrict__ A,    // [M,K]
    const unsigned short* __restrict__ Bt,   // [N,K]
    const float* __restrict__ bias,
    float* __restrict__ Cf,
    unsigned short* __restrict__ Cb,
    int M, int N, int K)
{
    // row stride 40 halfwords = 80B: 16B-aligned rows, bank-rotated (20 banks)
    __shared__ __align__(16) unsigned short sA[2][128 * 40];
    __shared__ __align__(16) unsigned short sB[2][256 * 40];
    const unsigned bufStrideA = 128 * 40 * 2;   // bytes between A buffers
    const unsigned bufStrideB = 256 * 40 * 2;   // bytes between B buffers

    const int t    = threadIdx.x;
    const int lane = t & 31;
    const int wave = t >> 5;
    const int wm   = wave >> 2;          // 0..1
    const int wn   = wave & 3;           // 0..3
    const int wM   = wm * 64;
    const int wN   = wn * 64;
    const int rowBase = blockIdx.y * 128;
    const int colBase = blockIdx.x * 256;
    const int m16  = lane & 15;
    const int hi   = lane >> 4;

    // Staging: A tile 128x32 halfwords = 512 x 16B (2/thread);
    //          B tile 256x32 halfwords = 1024 x 16B (4/thread).
    const int sr = t >> 2;
    const int cv = t & 3;
    const unsigned short* gA0 = A  + (size_t)(rowBase + sr) * K + cv * 8;
    const unsigned short* gA1 = gA0 + (size_t)64 * K;
    const unsigned short* gB0 = Bt + (size_t)(colBase + sr) * K + cv * 8;
    const unsigned short* gB1 = gB0 + (size_t)64 * K;
    const unsigned short* gB2 = gB0 + (size_t)128 * K;
    const unsigned short* gB3 = gB0 + (size_t)192 * K;
    const unsigned ldsA0 = (unsigned)(size_t)&sA[0][sr * 40 + cv * 8];
    const unsigned ldsA1 = ldsA0 + 64 * 40 * 2;
    const unsigned ldsB0 = (unsigned)(size_t)&sB[0][sr * 40 + cv * 8];
    const unsigned ldsB1 = ldsB0 + 64 * 40 * 2;
    const unsigned ldsB2 = ldsB0 + 128 * 40 * 2;
    const unsigned ldsB3 = ldsB0 + 192 * 40 * 2;

    v8f acc[4][4];
    #pragma unroll
    for (int i = 0; i < 4; ++i)
        #pragma unroll
        for (int j = 0; j < 4; ++j)
            #pragma unroll
            for (int e = 0; e < 8; ++e) acc[i][j][e] = 0.0f;

    // prologue: stage k-block 0 into buffer 0
    async_b128(ldsA0, gA0);
    async_b128(ldsA1, gA1);
    async_b128(ldsB0, gB0);
    async_b128(ldsB1, gB1);
    async_b128(ldsB2, gB2);
    async_b128(ldsB3, gB3);

    const int nk = K >> 5;
    for (int i = 0; i < nk; ++i) {
        const int buf = i & 1;
        if (i + 1 < nk) {
            const int k0 = (i + 1) << 5;
            const unsigned offA = (buf ^ 1) * bufStrideA;
            const unsigned offB = (buf ^ 1) * bufStrideB;
            async_b128(ldsA0 + offA, gA0 + k0);
            async_b128(ldsA1 + offA, gA1 + k0);
            async_b128(ldsB0 + offB, gB0 + k0);
            async_b128(ldsB1 + offB, gB1 + k0);
            async_b128(ldsB2 + offB, gB2 + k0);
            async_b128(ldsB3 + offB, gB3 + k0);
            wait_async_le6();     // oldest stage (ours) complete
        } else {
            wait_async_0();
        }
        __syncthreads();          // all waves' stage-i data visible in LDS

        const unsigned short* sAb = sA[buf];
        const unsigned short* sBb = sB[buf];

        // B fragments: lane n = lane&15; elements 0..15 hold K = 16*hi + e
        FragB16 fb[4];
        #pragma unroll
        for (int nt = 0; nt < 4; ++nt) {
            const unsigned short* bp = &sBb[(wN + nt * 16 + m16) * 40 + 16 * hi];
            fb[nt].q[0] = *(const u32x4*)bp;
            fb[nt].q[1] = *(const u32x4*)(bp + 8);
        }
        // A fragments + 16 WMMAs
        #pragma unroll
        for (int mt = 0; mt < 4; ++mt) {
            FragB16 fa;
            const unsigned short* ap = &sAb[(wM + mt * 16 + m16) * 40 + 16 * hi];
            fa.q[0] = *(const u32x4*)ap;          // K = 16*hi .. 16*hi+7
            fa.q[1] = *(const u32x4*)(ap + 8);    // K = 16*hi+8 .. +7
            #pragma unroll
            for (int nt = 0; nt < 4; ++nt) {
                acc[mt][nt] = __builtin_amdgcn_wmma_f32_16x16x32_bf16(
                    false, fa.v, false, fb[nt].v,
                    (short)0, acc[mt][nt], false, false);
            }
        }
        __syncthreads();          // safe to overwrite this buffer next stage
    }

    // epilogue: C tile layout: lane (hi,n); VGPR r -> (M = 8*hi + r, N = n)
    #pragma unroll
    for (int mt = 0; mt < 4; ++mt) {
        #pragma unroll
        for (int nt = 0; nt < 4; ++nt) {
            int col = colBase + wN + nt * 16 + m16;
            float bv = bias[col];
            #pragma unroll
            for (int r = 0; r < 8; ++r) {
                int row = rowBase + wM + mt * 16 + hi * 8 + r;
                float v = acc[mt][nt][r] + bv;
                if (RELU) v = fmaxf(v, 0.0f);
                if (OUT_BF16) Cb[(size_t)row * N + col] = f32_to_bf16(v);
                else          Cf[(size_t)row * N + col] = v;
            }
        }
    }
}

// ---------------------------------------------------------------------------
// Linear-attention ctx (pointwise over head-depth): one wave per (token, head)
// q = elu(qpre)+1, k = elu(kpre)+1; ks = sum_d k; kv = sum_d k*v
// ctx_d = q_d * kv / (q_d * ks + 1e-6)
// ---------------------------------------------------------------------------
__global__ __launch_bounds__(256) void attn_ctx_kernel(
    const unsigned short* __restrict__ qb,
    const unsigned short* __restrict__ kb,
    const unsigned short* __restrict__ vb,
    unsigned short* __restrict__ ctx,
    int total /* MROWS*HEADS */)
{
    int wave = (int)((blockIdx.x * 256 + threadIdx.x) >> 5);
    int lane = threadIdx.x & 31;
    if (wave >= total) return;
    int tok = wave >> 3;
    int h   = wave & 7;
    size_t base = (size_t)tok * D_MODEL + h * DHEAD + lane * 2;

    unsigned int qu = *(const unsigned int*)(qb + base);
    unsigned int ku = *(const unsigned int*)(kb + base);
    unsigned int vu = *(const unsigned int*)(vb + base);
    float q0 = bf16_to_f32(qu), q1 = bf16_to_f32(qu >> 16);
    float k0 = bf16_to_f32(ku), k1 = bf16_to_f32(ku >> 16);
    float v0 = bf16_to_f32(vu), v1 = bf16_to_f32(vu >> 16);

    // elu(x)+1
    q0 = q0 > 0.f ? q0 + 1.f : __expf(q0);
    q1 = q1 > 0.f ? q1 + 1.f : __expf(q1);
    k0 = k0 > 0.f ? k0 + 1.f : __expf(k0);
    k1 = k1 > 0.f ? k1 + 1.f : __expf(k1);

    float ks = k0 + k1;
    float kv = k0 * v0 + k1 * v1;
    #pragma unroll
    for (int off = 16; off > 0; off >>= 1) {
        ks += __shfl_xor(ks, off, 32);
        kv += __shfl_xor(kv, off, 32);
    }
    float c0 = q0 * kv / (q0 * ks + 1e-6f);
    float c1 = q1 * kv / (q1 * ks + 1e-6f);
    unsigned int cu = (unsigned int)f32_to_bf16(c0) |
                      ((unsigned int)f32_to_bf16(c1) << 16);
    *(unsigned int*)(ctx + base) = cu;
}

// ---------------------------------------------------------------------------
// Residual add + LayerNorm over 512; one wave per row; writes f32 + bf16.
// May run in place (xf_out == x): each wave owns its full row in registers.
// ---------------------------------------------------------------------------
__global__ __launch_bounds__(256) void add_ln_kernel(
    const float* x, const float* delta,
    const float* __restrict__ g, const float* __restrict__ b,
    float* xf_out, unsigned short* __restrict__ xb_out,
    int nrows)
{
    int row  = (int)((blockIdx.x * 256 + threadIdx.x) >> 5);
    int lane = threadIdx.x & 31;
    if (row >= nrows) return;
    size_t base = (size_t)row * D_MODEL;

    float s[16];
    float sum = 0.f;
    #pragma unroll
    for (int i = 0; i < 16; ++i) {
        int c = lane + i * 32;
        float v = x[base + c] + delta[base + c];
        s[i] = v; sum += v;
    }
    #pragma unroll
    for (int off = 16; off > 0; off >>= 1) sum += __shfl_xor(sum, off, 32);
    float mean = sum * (1.0f / D_MODEL);

    float var = 0.f;
    #pragma unroll
    for (int i = 0; i < 16; ++i) { float d = s[i] - mean; var += d * d; }
    #pragma unroll
    for (int off = 16; off > 0; off >>= 1) var += __shfl_xor(var, off, 32);
    float rs = rsqrtf(var * (1.0f / D_MODEL) + 1e-3f);

    #pragma unroll
    for (int i = 0; i < 16; ++i) {
        int c = lane + i * 32;
        float y = (s[i] - mean) * rs * g[c] + b[c];
        xf_out[base + c] = y;
        xb_out[base + c] = f32_to_bf16(y);
    }
}

// ---------------------------------------------------------------------------
// Host side
// ---------------------------------------------------------------------------
extern "C" void kernel_launch(void* const* d_in, const int* in_sizes, int n_in,
                              void* d_out, int out_size, void* d_ws, size_t ws_size,
                              hipStream_t stream) {
    (void)in_sizes; (void)n_in; (void)out_size; (void)ws_size;
    const int*   tokens = (const int*)  d_in[0];
    const float* emb    = (const float*)d_in[1];
    const float* pe     = (const float*)d_in[2];
    const float* Wq = (const float*)d_in[3];  const float* bq  = (const float*)d_in[4];
    const float* Wk = (const float*)d_in[5];  const float* bk  = (const float*)d_in[6];
    const float* Wv = (const float*)d_in[7];  const float* bv  = (const float*)d_in[8];
    const float* Wo = (const float*)d_in[9];  const float* bo  = (const float*)d_in[10];
    const float* g1 = (const float*)d_in[11]; const float* be1 = (const float*)d_in[12];
    const float* W1 = (const float*)d_in[13]; const float* b1  = (const float*)d_in[14];
    const float* W2 = (const float*)d_in[15]; const float* b2  = (const float*)d_in[16];
    const float* g2 = (const float*)d_in[17]; const float* be2 = (const float*)d_in[18];

    const size_t DD  = (size_t)D_MODEL * D_MODEL;    // 262144
    const size_t DF  = (size_t)D_MODEL * DFF;        // 1048576
    const size_t MD  = (size_t)MROWS * D_MODEL;      // 8388608
    const size_t MF  = (size_t)MROWS * DFF;          // 33554432

    char* p = (char*)d_ws;
    auto alloc = [&](size_t bytes) -> char* {
        char* r = p; p += (bytes + 255) & ~(size_t)255; return r;
    };
    unsigned short* wqB = (unsigned short*)alloc(LAYERS * DD * 2);  // [N,K] bf16
    unsigned short* wkB = (unsigned short*)alloc(LAYERS * DD * 2);
    unsigned short* wvB = (unsigned short*)alloc(LAYERS * DD * 2);
    unsigned short* woB = (unsigned short*)alloc(LAYERS * DD * 2);
    unsigned short* w1B = (unsigned short*)alloc(LAYERS * DF * 2);
    unsigned short* w2B = (unsigned short*)alloc(LAYERS * DF * 2);
    float*          x_f = (float*)         alloc(MD * 4);
    unsigned short* x_b = (unsigned short*)alloc(MD * 2);
    unsigned short* q_b = (unsigned short*)alloc(MD * 2);
    unsigned short* k_b = (unsigned short*)alloc(MD * 2);
    unsigned short* v_b = (unsigned short*)alloc(MD * 2);
    unsigned short* c_b = (unsigned short*)alloc(MD * 2);
    float*          t_f = (float*)         alloc(MD * 4);
    unsigned short* h_b = (unsigned short*)alloc(MF * 2);

    auto cvtT = [&](const float* src, unsigned short* dst, int K, int N) {
        size_t n = (size_t)LAYERS * K * N;
        cvtT_bf16_kernel<<<(unsigned)((n + 255) / 256), 256, 0, stream>>>(src, dst, K, N, n);
    };
    cvtT(Wq, wqB, D_MODEL, D_MODEL);
    cvtT(Wk, wkB, D_MODEL, D_MODEL);
    cvtT(Wv, wvB, D_MODEL, D_MODEL);
    cvtT(Wo, woB, D_MODEL, D_MODEL);
    cvtT(W1, w1B, D_MODEL, DFF);     // -> [DFF, D_MODEL]
    cvtT(W2, w2B, DFF, D_MODEL);     // -> [D_MODEL, DFF]

    embed_kernel<<<(unsigned)(MD / 256), 256, 0, stream>>>(tokens, emb, pe, x_f, x_b);

    const dim3 blk(256);
    for (int i = 0; i < LAYERS; ++i) {
        const unsigned short* wq = wqB + (size_t)i * DD;
        const unsigned short* wk = wkB + (size_t)i * DD;
        const unsigned short* wv = wvB + (size_t)i * DD;
        const unsigned short* wo = woB + (size_t)i * DD;
        const unsigned short* w1 = w1B + (size_t)i * DF;
        const unsigned short* w2 = w2B + (size_t)i * DF;

        dim3 gp(D_MODEL / 256, MROWS / 128);   // 2 x 128
        // Q, K, V projections (bf16 out; elu applied later)
        gemm_bf16_wmma<1, 0><<<gp, blk, 0, stream>>>(x_b, wq, bq + (size_t)i * D_MODEL,
                                                     nullptr, q_b, MROWS, D_MODEL, D_MODEL);
        gemm_bf16_wmma<1, 0><<<gp, blk, 0, stream>>>(x_b, wk, bk + (size_t)i * D_MODEL,
                                                     nullptr, k_b, MROWS, D_MODEL, D_MODEL);
        gemm_bf16_wmma<1, 0><<<gp, blk, 0, stream>>>(x_b, wv, bv + (size_t)i * D_MODEL,
                                                     nullptr, v_b, MROWS, D_MODEL, D_MODEL);
        // pointwise linear-attention ctx
        attn_ctx_kernel<<<(unsigned)(MROWS * HEADS / 8), blk, 0, stream>>>(
            q_b, k_b, v_b, c_b, MROWS * HEADS);
        // output projection (f32 out)
        gemm_bf16_wmma<0, 0><<<gp, blk, 0, stream>>>(c_b, wo, bo + (size_t)i * D_MODEL,
                                                     t_f, nullptr, MROWS, D_MODEL, D_MODEL);
        // x = LN(x + attn)
        add_ln_kernel<<<(unsigned)(MROWS / 8), blk, 0, stream>>>(
            x_f, t_f, g1 + (size_t)i * D_MODEL, be1 + (size_t)i * D_MODEL,
            x_f, x_b, MROWS);
        // FFN1: relu(x @ W1 + b1) -> bf16
        dim3 g1d(DFF / 256, MROWS / 128);      // 8 x 128
        gemm_bf16_wmma<1, 1><<<g1d, blk, 0, stream>>>(x_b, w1, b1 + (size_t)i * DFF,
                                                      nullptr, h_b, MROWS, DFF, D_MODEL);
        // FFN2: h @ W2 + b2 -> f32
        gemm_bf16_wmma<0, 0><<<gp, blk, 0, stream>>>(h_b, w2, b2 + (size_t)i * D_MODEL,
                                                     t_f, nullptr, MROWS, D_MODEL, DFF);
        // x = LN(x + ffn); last layer writes straight to d_out
        float* xout = (i == LAYERS - 1) ? (float*)d_out : x_f;
        add_ln_kernel<<<(unsigned)(MROWS / 8), blk, 0, stream>>>(
            x_f, t_f, g2 + (size_t)i * D_MODEL, be2 + (size_t)i * D_MODEL,
            xout, x_b, MROWS);
    }
}
